// ContrastiveLinearLoss_3109556322832
// MI455X (gfx1250) — matrix operand order repaired and checked by reading, not compiled, not run on hardware
//
#include <hip/hip_runtime.h>
#include <hip/hip_bf16.h>
#include <stdint.h>

// Problem constants (match reference)
#define NROWS 4096
#define DCOLS 4096
#define MARGIN 0.5f
#define EPS 1e-8f

// Tiling
#define MT 128                       // macro tile (rows & cols)
#define NB (NROWS / MT)              // 32 macro tiles per dim
#define NPAIR (NB * (NB + 1) / 2)    // 528 upper-tri macro tile pairs
#define KSTEP 32                     // K depth staged per panel (one WMMA K)
#define NSTEP (DCOLS / KSTEP)        // 128 k-steps
#define LDA 40                       // padded LDS row stride (bf16): 80B, 16B-aligned, bank-friendly
#define PANEL (MT * LDA)             // elements per LDS panel buffer

typedef __attribute__((ext_vector_type(16))) __bf16 v16bf;
typedef __attribute__((ext_vector_type(8)))  float  v8f;

// ---------------------------------------------------------------------------
// Kernel 1: row-normalize f32 -> bf16 (RNE), wn[i][j] = w[i][j]/max(||w_i||,eps)
// ---------------------------------------------------------------------------
__global__ void __launch_bounds__(256) rownorm_bf16(const float* __restrict__ w,
                                                    uint16_t* __restrict__ out) {
    const int row = blockIdx.x;
    const float* r = w + (size_t)row * DCOLS;
    __shared__ float red[256];

    float ss = 0.0f;
    for (int c = threadIdx.x; c < DCOLS; c += 256) {
        float v = r[c];
        ss += v * v;
    }
    red[threadIdx.x] = ss;
    __syncthreads();
    for (int s = 128; s > 0; s >>= 1) {
        if (threadIdx.x < s) red[threadIdx.x] += red[threadIdx.x + s];
        __syncthreads();
    }
    const float scale = 1.0f / fmaxf(sqrtf(red[0]), EPS);

    uint16_t* o = out + (size_t)row * DCOLS;
    for (int c = threadIdx.x; c < DCOLS; c += 256) {
        float v = r[c] * scale;
        uint32_t b = __float_as_uint(v);
        uint32_t r16 = (b + 0x7FFFu + ((b >> 16) & 1u)) >> 16;  // RNE f32->bf16
        o[c] = (uint16_t)r16;
    }
}

// ---------------------------------------------------------------------------
// Kernel 2: triangular tiled GEMM (wn @ wn^T) + hinge + per-block reduction.
// Block = 256 threads (8 waves) computes a 128x128 sim macro-tile.
// Waves 4x2: wave (wr,wc) owns rows [wr*32,+32) x cols [wc*64,+64)
// => 2 A frags, 4 B frags, 8 WMMA accumulators per wave per k-step.
// Double-buffered LDS panels with toggling pointers; steady loop is
// NOT unrolled (keeps accumulators in fixed register tuples); last
// k-step peeled (no prefetch branch in the hot body).
// ---------------------------------------------------------------------------
__global__ void __launch_bounds__(256, 1) tri_hinge_gemm(const uint16_t* __restrict__ wn,
                                                         float* __restrict__ partials) {
    // Map linear block id -> (bi, bj) with bj >= bi (upper triangle of macro tiles)
    int t = blockIdx.x;
    int bi = 0;
    int rowlen = NB;
    while (t >= rowlen) { t -= rowlen; rowlen--; bi++; }
    const int bj = bi + t;
    const int i0 = bi * MT;
    const int j0 = bj * MT;

    __shared__ __align__(16) uint16_t As[2][PANEL];
    __shared__ __align__(16) uint16_t Bs[2][PANEL];
    __shared__ float rsum[256];
    __shared__ float rcnt[256];

    const int tid  = threadIdx.x;
    const int wave = tid >> 5;      // 0..7
    const int lane = tid & 31;
    const int l16  = lane & 15;
    const int half = lane >> 4;     // 0: lanes 0-15, 1: lanes 16-31
    const int wr   = wave >> 1;     // 0..3  (row group of 32)
    const int wc   = wave & 1;      // 0..1  (col group of 64)

    v8f acc[2][4] = {};

    // Cooperative global->LDS staging: thread t moves one 32B segment per panel.
    const int ldrow = tid >> 1;     // 0..127
    const int ldseg = tid & 1;      // which 16-bf16 half of the 32-col panel
    const int ldoff = ldrow * LDA + ldseg * 16;
    const size_t a_gbase = (size_t)(i0 + ldrow) * DCOLS + (size_t)ldseg * 16;
    const size_t b_gbase = (size_t)(j0 + ldrow) * DCOLS + (size_t)ldseg * 16;

    // Fragment read offsets within a panel (a=1 tile adds 16*LDA as imm offset;
    // B tile c adds c*16*LDA as imm offset).
    const int a_read_off = (wr * 32 + l16) * LDA + half * 16;  // + {0,8}  for K-chunks
    const int b_read_off = (wc * 64 + l16) * LDA + half * 16;  // + {0,8}  for K-chunks

    // Toggling read/write pointers (read buf0 first, write buf1 first).
    const uint16_t* ar = &As[0][a_read_off];
    const uint16_t* br = &Bs[0][b_read_off];
    uint16_t* aw = &As[1][ldoff];
    uint16_t* bw = &Bs[1][ldoff];
    int delta = PANEL;

    // Preload panel 0 into buffer 0
    {
        const uint4* ag = (const uint4*)(wn + a_gbase);
        const uint4* bg = (const uint4*)(wn + b_gbase);
        uint4 a0 = ag[0], a1 = ag[1], b0 = bg[0], b1 = bg[1];
        uint4* al = (uint4*)&As[0][ldoff];
        uint4* bl = (uint4*)&Bs[0][ldoff];
        al[0] = a0; al[1] = a1;
        bl[0] = b0; bl[1] = b1;
    }
    __syncthreads();

    union Frag { uint4 u[2]; v16bf v; };

#pragma clang loop unroll(disable)
    for (int s = 0; s < NSTEP - 1; ++s) {
        // Issue next panel's global loads early (hide latency under WMMAs)
        const int k = (s + 1) * KSTEP;
        const uint4* ag = (const uint4*)(wn + a_gbase + k);
        const uint4* bg = (const uint4*)(wn + b_gbase + k);
        uint4 na0 = ag[0], na1 = ag[1], nb0 = bg[0], nb1 = bg[1];

        // Load ALL fragments into distinct registers, then the WMMA burst.
        // A frag (16x32 bf16): lanes 0-15 K 0..15, lanes 16-31 K 16..31
        Frag af0, af1, bf0, bf1, bf2, bf3;
        af0.u[0] = *(const uint4*)(ar);
        af0.u[1] = *(const uint4*)(ar + 8);
        af1.u[0] = *(const uint4*)(ar + 16 * LDA);
        af1.u[1] = *(const uint4*)(ar + 16 * LDA + 8);
        // B frag (32x16 bf16): lanes 0-15 K 0..15, lanes 16-31 K 16..31
        bf0.u[0] = *(const uint4*)(br);
        bf0.u[1] = *(const uint4*)(br + 8);
        bf1.u[0] = *(const uint4*)(br + 16 * LDA);
        bf1.u[1] = *(const uint4*)(br + 16 * LDA + 8);
        bf2.u[0] = *(const uint4*)(br + 32 * LDA);
        bf2.u[1] = *(const uint4*)(br + 32 * LDA + 8);
        bf3.u[0] = *(const uint4*)(br + 48 * LDA);
        bf3.u[1] = *(const uint4*)(br + 48 * LDA + 8);

        acc[0][0] = __builtin_amdgcn_wmma_f32_16x16x32_bf16(false, af0.v, false, bf0.v, (short)0, acc[0][0], false, false);
        acc[0][1] = __builtin_amdgcn_wmma_f32_16x16x32_bf16(false, af0.v, false, bf1.v, (short)0, acc[0][1], false, false);
        acc[0][2] = __builtin_amdgcn_wmma_f32_16x16x32_bf16(false, af0.v, false, bf2.v, (short)0, acc[0][2], false, false);
        acc[0][3] = __builtin_amdgcn_wmma_f32_16x16x32_bf16(false, af0.v, false, bf3.v, (short)0, acc[0][3], false, false);
        acc[1][0] = __builtin_amdgcn_wmma_f32_16x16x32_bf16(false, af1.v, false, bf0.v, (short)0, acc[1][0], false, false);
        acc[1][1] = __builtin_amdgcn_wmma_f32_16x16x32_bf16(false, af1.v, false, bf1.v, (short)0, acc[1][1], false, false);
        acc[1][2] = __builtin_amdgcn_wmma_f32_16x16x32_bf16(false, af1.v, false, bf2.v, (short)0, acc[1][2], false, false);
        acc[1][3] = __builtin_amdgcn_wmma_f32_16x16x32_bf16(false, af1.v, false, bf3.v, (short)0, acc[1][3], false, false);

        // Park next panel in the other LDS buffer; toggle pointers; one barrier.
        uint4* al = (uint4*)aw;
        uint4* bl = (uint4*)bw;
        al[0] = na0; al[1] = na1;
        bl[0] = nb0; bl[1] = nb1;

        ar += delta; br += delta;
        aw -= delta; bw -= delta;
        delta = -delta;
        __syncthreads();
    }

    // Peeled final k-step: compute only, no prefetch/store.
    {
        Frag af0, af1, bf0, bf1, bf2, bf3;
        af0.u[0] = *(const uint4*)(ar);
        af0.u[1] = *(const uint4*)(ar + 8);
        af1.u[0] = *(const uint4*)(ar + 16 * LDA);
        af1.u[1] = *(const uint4*)(ar + 16 * LDA + 8);
        bf0.u[0] = *(const uint4*)(br);
        bf0.u[1] = *(const uint4*)(br + 8);
        bf1.u[0] = *(const uint4*)(br + 16 * LDA);
        bf1.u[1] = *(const uint4*)(br + 16 * LDA + 8);
        bf2.u[0] = *(const uint4*)(br + 32 * LDA);
        bf2.u[1] = *(const uint4*)(br + 32 * LDA + 8);
        bf3.u[0] = *(const uint4*)(br + 48 * LDA);
        bf3.u[1] = *(const uint4*)(br + 48 * LDA + 8);

        acc[0][0] = __builtin_amdgcn_wmma_f32_16x16x32_bf16(false, af0.v, false, bf0.v, (short)0, acc[0][0], false, false);
        acc[0][1] = __builtin_amdgcn_wmma_f32_16x16x32_bf16(false, af0.v, false, bf1.v, (short)0, acc[0][1], false, false);
        acc[0][2] = __builtin_amdgcn_wmma_f32_16x16x32_bf16(false, af0.v, false, bf2.v, (short)0, acc[0][2], false, false);
        acc[0][3] = __builtin_amdgcn_wmma_f32_16x16x32_bf16(false, af0.v, false, bf3.v, (short)0, acc[0][3], false, false);
        acc[1][0] = __builtin_amdgcn_wmma_f32_16x16x32_bf16(false, af1.v, false, bf0.v, (short)0, acc[1][0], false, false);
        acc[1][1] = __builtin_amdgcn_wmma_f32_16x16x32_bf16(false, af1.v, false, bf1.v, (short)0, acc[1][1], false, false);
        acc[1][2] = __builtin_amdgcn_wmma_f32_16x16x32_bf16(false, af1.v, false, bf2.v, (short)0, acc[1][2], false, false);
        acc[1][3] = __builtin_amdgcn_wmma_f32_16x16x32_bf16(false, af1.v, false, bf3.v, (short)0, acc[1][3], false, false);
    }

    // Hinge + strict-upper-triangle mask + thread-local accumulation.
    // C layout: VGPR v, lanes 0-15: (M=v, N=lane); lanes 16-31: (M=v+8, N=lane-16)
    float lsum = 0.0f;
    float lcnt = 0.0f;
    const int gi_base = i0 + wr * 32 + (half ? 8 : 0);
    #pragma unroll
    for (int a = 0; a < 2; ++a) {
        #pragma unroll
        for (int c = 0; c < 4; ++c) {
            const int gj = j0 + wc * 64 + c * 16 + l16;
            #pragma unroll
            for (int v = 0; v < 8; ++v) {
                const int gi = gi_base + a * 16 + v;
                const float h = acc[a][c][v] - MARGIN;
                if ((gj > gi) && (h > 0.0f)) {
                    lsum += h;
                    lcnt += 1.0f;
                }
            }
        }
    }

    // Deterministic block reduction -> fixed partial slot (no float atomics).
    rsum[tid] = lsum;
    rcnt[tid] = lcnt;
    __syncthreads();
    for (int s = 128; s > 0; s >>= 1) {
        if (tid < s) {
            rsum[tid] += rsum[tid + s];
            rcnt[tid] += rcnt[tid + s];
        }
        __syncthreads();
    }
    if (tid == 0) {
        partials[(size_t)blockIdx.x * 2 + 0] = rsum[0];
        partials[(size_t)blockIdx.x * 2 + 1] = rcnt[0];
    }
}

// ---------------------------------------------------------------------------
// Kernel 3: fixed-order reduction of per-block partials -> scalar loss.
// ---------------------------------------------------------------------------
__global__ void __launch_bounds__(256) finalize_loss(const float* __restrict__ partials,
                                                     float* __restrict__ out) {
    __shared__ float s0[256], c0[256], s1[256], c1[256];
    const int tid = threadIdx.x;
    float a = 0.0f, b = 0.0f, cs = 0.0f, cc = 0.0f;
    for (int i = tid; i < NPAIR; i += 256) {
        a  += partials[2 * i + 0];
        b  += partials[2 * i + 1];
        cs += partials[2 * NPAIR + 2 * i + 0];
        cc += partials[2 * NPAIR + 2 * i + 1];
    }
    s0[tid] = a; c0[tid] = b; s1[tid] = cs; c1[tid] = cc;
    __syncthreads();
    for (int s = 128; s > 0; s >>= 1) {
        if (tid < s) {
            s0[tid] += s0[tid + s];
            c0[tid] += c0[tid + s];
            s1[tid] += s1[tid + s];
            c1[tid] += c1[tid + s];
        }
        __syncthreads();
    }
    if (tid == 0) {
        const float l0 = s0[0] / fmaxf(c0[0], 1.0f);
        const float l1 = s1[0] / fmaxf(c1[0], 1.0f);
        out[0] = 0.5f * (l0 + l1);
    }
}

// ---------------------------------------------------------------------------
// Launch. Workspace: [0,16KB) partials (2 layers x 528 x {sum,cnt}),
//                    [16KB, 16KB+32MB) normalized bf16 matrix (reused per layer).
// ---------------------------------------------------------------------------
extern "C" void kernel_launch(void* const* d_in, const int* in_sizes, int n_in,
                              void* d_out, int out_size, void* d_ws, size_t ws_size,
                              hipStream_t stream) {
    const float* w0 = (const float*)d_in[0];
    const float* w1 = (const float*)d_in[1];
    float* out = (float*)d_out;

    char* ws = (char*)d_ws;
    float* partials = (float*)ws;                        // 2 * NPAIR * 2 floats
    uint16_t* wn = (uint16_t*)(ws + 16384);              // 4096*4096 bf16 = 32MB

    // Layer 0
    rownorm_bf16<<<NROWS, 256, 0, stream>>>(w0, wn);
    tri_hinge_gemm<<<NPAIR, 256, 0, stream>>>(wn, partials);

    // Layer 1 (reuses wn buffer; same stream enforces ordering)
    rownorm_bf16<<<NROWS, 256, 0, stream>>>(w1, wn);
    tri_hinge_gemm<<<NPAIR, 256, 0, stream>>>(wn, partials + 2 * NPAIR);

    finalize_loss<<<1, 256, 0, stream>>>(partials, out);
}
